// HybridBranch_62981400429060
// MI455X (gfx1250) — matrix operand order repaired and checked by reading, not compiled, not run on hardware
//
#include <hip/hip_runtime.h>
#include <hip/hip_bf16.h>

// ---------------------------------------------------------------------------
// Problem constants (from reference)
// ---------------------------------------------------------------------------
#define Bv        256
#define HID       512
#define NSTEP     32
#define NCLS      97
#define Lspat     256          // H*W = 8*32
#define G4        2048         // 4*HID
#define KCAT      1024         // 2*HID
#define ROWS      (Bv*NSTEP)   // 8192
#define NGEN      128          // 97 padded to 128 (multiple of 64)

// WMMA vector types (CDNA5 / gfx1250, wave32)
typedef __attribute__((ext_vector_type(16))) __bf16 v16bf;
typedef __attribute__((ext_vector_type(8)))  __bf16 v8bf;
typedef __attribute__((ext_vector_type(8)))  float  v8f;

union FragBF { v16bf v; v8bf h[2]; };

__device__ __forceinline__ unsigned short f2bf(float f) {
    unsigned int u = __float_as_uint(f);
    unsigned int r = u + 0x7FFFu + ((u >> 16) & 1u);   // round-to-nearest-even
    return (unsigned short)(r >> 16);
}
__device__ __forceinline__ float sigm(float x) { return 1.f / (1.f + __expf(-x)); }

// ---------------------------------------------------------------------------
// One-time: pack [W_ih | W_hh] fp32 -> bf16, concatenated along K (row-major NxK)
// ---------------------------------------------------------------------------
__global__ void pack_lstm_weights(const float* __restrict__ Wih0, const float* __restrict__ Whh0,
                                  const float* __restrict__ Wih1, const float* __restrict__ Whh1,
                                  unsigned short* __restrict__ wcat0, unsigned short* __restrict__ wcat1) {
    int gid = blockIdx.x * 256 + threadIdx.x;          // over 2048*1024
    int n = gid >> 10, k = gid & 1023;
    float w0 = (k < HID) ? Wih0[n * HID + k] : Whh0[n * HID + (k - HID)];
    float w1 = (k < HID) ? Wih1[n * HID + k] : Whh1[n * HID + (k - HID)];
    wcat0[gid] = f2bf(w0);
    wcat1[gid] = f2bf(w1);
}

// One-time: gen_W [97,512] fp32 -> bf16 padded to [128,512]
__global__ void pack_gen_weights(const float* __restrict__ genW, unsigned short* __restrict__ genWbf) {
    int gid = blockIdx.x * 256 + threadIdx.x;          // over 128*512
    int n = gid >> 9, k = gid & 511;
    float v = (n < NCLS) ? genW[n * HID + k] : 0.f;
    genWbf[gid] = f2bf(v);
}

// Per step: x_emb = emb_W[:, idx] + emb_b  ->  bf16 into first half of A0 (stride 1024)
__global__ void emb_fill(const int* __restrict__ text, const float* __restrict__ embW,
                         const float* __restrict__ embB, unsigned short* __restrict__ A0, int t) {
    int gid = blockIdx.x * 256 + threadIdx.x;          // over 256*512
    int b = gid >> 9, h = gid & 511;
    int idx = text[b * NSTEP + t];
    float v = embW[h * NCLS + idx] + embB[h];
    A0[(size_t)b * KCAT + h] = f2bf(v);
}

// ---------------------------------------------------------------------------
// bf16 WMMA GEMM:  D[M x N] = A[M x K] * W[N x K]^T   (f32 accumulate)
// One wave -> 32x64 output block: 2 A fragments x 4 B fragments -> 8
// v_wmma_f32_16x16x32_bf16 per K-step against 12 b128 loads (each B fragment
// amortized over two WMMAs; halves L2 weight traffic vs a 16x64 strip).
// A/B fragments use contiguous K per lane half (same order in both); D layout per CDNA5 ISA 7.12.2 (wave32).
// ---------------------------------------------------------------------------
__global__ void gemm_bf16_wmma(const unsigned short* __restrict__ A_, const unsigned short* __restrict__ W_,
                               float* __restrict__ D, int M32, int N64, int K,
                               int lda, int ldw, int ldd) {
    int wid  = (blockIdx.x * blockDim.x + threadIdx.x) >> 5;
    int lane = threadIdx.x & 31;
    int wm = wid % M32;
    int wn = wid / M32;
    if (wn >= N64) return;

    const __bf16* A = reinterpret_cast<const __bf16*>(A_);
    const __bf16* W = reinterpret_cast<const __bf16*>(W_);

    // A fragment: lanes 0-15 -> row (lane), K 0..15; lanes 16-31 -> row (lane-16), K 16..31 (contiguous)
    int rowA = wm * 32 + (lane & 15);
    int aoff = (lane >= 16) ? 16 : 0;
    const __bf16* Ap0 = A + (size_t)rowA * lda + aoff;          // rows 0..15 of strip
    const __bf16* Ap1 = A + (size_t)(rowA + 16) * lda + aoff;   // rows 16..31 of strip

    // B fragment: lane column = lane&15 ; lanes 0-15 -> K 0..15, lanes 16-31 -> K 16..31 (contiguous)
    int ncol = lane & 15;
    int boff = (lane >= 16) ? 16 : 0;
    const __bf16* Bp0 = W + (size_t)(wn * 64 +  0 + ncol) * ldw + boff;
    const __bf16* Bp1 = W + (size_t)(wn * 64 + 16 + ncol) * ldw + boff;
    const __bf16* Bp2 = W + (size_t)(wn * 64 + 32 + ncol) * ldw + boff;
    const __bf16* Bp3 = W + (size_t)(wn * 64 + 48 + ncol) * ldw + boff;

    v8f c00 = {}, c01 = {}, c02 = {}, c03 = {};   // rows 0..15   x 4 col tiles
    v8f c10 = {}, c11 = {}, c12 = {}, c13 = {};   // rows 16..31  x 4 col tiles

    for (int k0 = 0; k0 < K; k0 += 32) {
        // keep the weight stream ahead of WMMA consumption (global_prefetch_b8)
        __builtin_prefetch(Bp0 + k0 + 256, 0, 3);
        __builtin_prefetch(Bp1 + k0 + 256, 0, 3);
        __builtin_prefetch(Bp2 + k0 + 256, 0, 3);
        __builtin_prefetch(Bp3 + k0 + 256, 0, 3);

        FragBF a0, a1;
        a0.h[0] = *reinterpret_cast<const v8bf*>(Ap0 + k0);
        a0.h[1] = *reinterpret_cast<const v8bf*>(Ap0 + k0 + 8);
        a1.h[0] = *reinterpret_cast<const v8bf*>(Ap1 + k0);
        a1.h[1] = *reinterpret_cast<const v8bf*>(Ap1 + k0 + 8);
        v16bf b0 = *reinterpret_cast<const v16bf*>(Bp0 + k0);
        v16bf b1 = *reinterpret_cast<const v16bf*>(Bp1 + k0);
        v16bf b2 = *reinterpret_cast<const v16bf*>(Bp2 + k0);
        v16bf b3 = *reinterpret_cast<const v16bf*>(Bp3 + k0);

        c00 = __builtin_amdgcn_wmma_f32_16x16x32_bf16(false, a0.v, false, b0, (short)0, c00, false, false);
        c10 = __builtin_amdgcn_wmma_f32_16x16x32_bf16(false, a1.v, false, b0, (short)0, c10, false, false);
        c01 = __builtin_amdgcn_wmma_f32_16x16x32_bf16(false, a0.v, false, b1, (short)0, c01, false, false);
        c11 = __builtin_amdgcn_wmma_f32_16x16x32_bf16(false, a1.v, false, b1, (short)0, c11, false, false);
        c02 = __builtin_amdgcn_wmma_f32_16x16x32_bf16(false, a0.v, false, b2, (short)0, c02, false, false);
        c12 = __builtin_amdgcn_wmma_f32_16x16x32_bf16(false, a1.v, false, b2, (short)0, c12, false, false);
        c03 = __builtin_amdgcn_wmma_f32_16x16x32_bf16(false, a0.v, false, b3, (short)0, c03, false, false);
        c13 = __builtin_amdgcn_wmma_f32_16x16x32_bf16(false, a1.v, false, b3, (short)0, c13, false, false);
    }

    // D layout: VGPR v -> row (v or v+8); lanes 0-15 cols N, lanes 16-31 rows +8
    int rBase = wm * 32 + ((lane >= 16) ? 8 : 0);
    int cBase = wn * 64 + ncol;
#pragma unroll
    for (int v = 0; v < 8; ++v) {
        size_t r0 = (size_t)(rBase + v) * ldd;
        size_t r1 = (size_t)(rBase + 16 + v) * ldd;
        D[r0 + cBase +  0] = c00[v];
        D[r0 + cBase + 16] = c01[v];
        D[r0 + cBase + 32] = c02[v];
        D[r0 + cBase + 48] = c03[v];
        D[r1 + cBase +  0] = c10[v];
        D[r1 + cBase + 16] = c11[v];
        D[r1 + cBase + 32] = c12[v];
        D[r1 + cBase + 48] = c13[v];
    }
}

// ---------------------------------------------------------------------------
// LSTM elementwise: gates [B,2048] (i,f,g,o) + biases -> h,c (f32) + h in bf16
// written into the A-buffers (stride KCAT) for the next GEMMs.
// ---------------------------------------------------------------------------
__global__ void lstm_ew(const float* __restrict__ gates, const float* __restrict__ bih,
                        const float* __restrict__ bhh, float* __restrict__ h, float* __restrict__ c,
                        unsigned short* __restrict__ dst1, unsigned short* __restrict__ dst2) {
    int gid = blockIdx.x * 256 + threadIdx.x;          // over 256*512
    int b = gid >> 9, j = gid & 511;
    const float* gr = gates + (size_t)b * G4;
    float i_ = gr[j]            + bih[j]            + bhh[j];
    float f_ = gr[HID + j]      + bih[HID + j]      + bhh[HID + j];
    float g_ = gr[2 * HID + j]  + bih[2 * HID + j]  + bhh[2 * HID + j];
    float o_ = gr[3 * HID + j]  + bih[3 * HID + j]  + bhh[3 * HID + j];
    float cn = sigm(f_) * c[gid] + sigm(i_) * tanhf(g_);
    float hn = sigm(o_) * tanhf(cn);
    c[gid] = cn;
    h[gid] = hn;
    unsigned short hb = f2bf(hn);
    dst1[(size_t)b * KCAT + j] = hb;
    if (dst2) dst2[(size_t)b * KCAT + j] = hb;
}

// ---------------------------------------------------------------------------
// Fused attention + context, one block (256 thr = 8 waves) per batch element.
//   a[b,l]      = sigmoid( sum_ch h1[b,ch] * origin[b,ch,l] )   (coalesced over l)
//   context[b,c]= sum_l a[b,l] * origin[b,c,l]                   (wave-reduced)
// fmap rows live in L2 (128 MB < 192 MB), so re-reading per step is cheap.
// ---------------------------------------------------------------------------
__global__ void attn_ctx(const float* __restrict__ origin, const float* __restrict__ h1,
                         float* __restrict__ masks, float* __restrict__ hiddens,
                         unsigned short* __restrict__ ctxbf, int t) {
    int b = blockIdx.x, tid = threadIdx.x;
    __shared__ float h1s[HID];
    __shared__ float av[Lspat];
    const float* fb = origin + (size_t)b * HID * Lspat;
    h1s[tid]       = h1[b * HID + tid];
    h1s[tid + 256] = h1[b * HID + 256 + tid];
    __syncthreads();

    float acc = 0.f;
    for (int ch = 0; ch < HID; ++ch)
        acc += h1s[ch] * fb[(size_t)ch * Lspat + tid];
    float a = sigm(acc);
    masks[(size_t)b * (NSTEP * Lspat) + t * Lspat + tid] = a;
    av[tid] = a;
    __syncthreads();

    int wave = tid >> 5, lane = tid & 31;
    const float* ap = av + lane * 8;
    for (int j = 0; j < 64; ++j) {
        int ch = wave * 64 + j;
        const float* row = fb + (size_t)ch * Lspat + lane * 8;
        float p = 0.f;
#pragma unroll
        for (int e = 0; e < 8; ++e) p += ap[e] * row[e];
#pragma unroll
        for (int off = 16; off > 0; off >>= 1) p += __shfl_xor(p, off, 32);
        if (lane == 0) {
            size_t r = (size_t)(b * NSTEP + t);
            hiddens[r * HID + ch] = p;           // output_hiddens (fp32 out)
            ctxbf[r * HID + ch] = f2bf(p);       // bf16 copy for generator GEMM
        }
    }
}

// Generator epilogue: g = Dgen[:, :97] + gen_b
__global__ void gen_epilogue(const float* __restrict__ dgen, const float* __restrict__ genB,
                             float* __restrict__ g) {
    int gid = blockIdx.x * 256 + threadIdx.x;
    if (gid >= ROWS * NCLS) return;
    int row = gid / NCLS, n = gid % NCLS;
    g[gid] = dgen[(size_t)row * NGEN + n] + genB[n];
}

// ---------------------------------------------------------------------------
// Workspace layout (bytes)
// ---------------------------------------------------------------------------
#define OFF_WCAT0   0u                                     // 2048*1024 bf16 = 4 MiB
#define OFF_WCAT1   (OFF_WCAT0 + 2048u*1024u*2u)
#define OFF_GENW    (OFF_WCAT1 + 2048u*1024u*2u)           // 128*512 bf16
#define OFF_A0      (OFF_GENW  + 128u*512u*2u)             // [256,1024] bf16
#define OFF_A1      (OFF_A0    + 256u*1024u*2u)
#define OFF_H0      (OFF_A1    + 256u*1024u*2u)            // 4 x [256,512] f32
#define OFF_C0      (OFF_H0    + 256u*512u*4u)
#define OFF_H1      (OFF_C0    + 256u*512u*4u)
#define OFF_C1      (OFF_H1    + 256u*512u*4u)
#define OFF_GATES   (OFF_C1    + 256u*512u*4u)             // [256,2048] f32
#define OFF_CTXBF   (OFF_GATES + 256u*2048u*4u)            // [8192,512] bf16
#define OFF_DGEN    (OFF_CTXBF + 8192u*512u*2u)            // [8192,128] f32

extern "C" void kernel_launch(void* const* d_in, const int* in_sizes, int n_in,
                              void* d_out, int out_size, void* d_ws, size_t ws_size,
                              hipStream_t stream) {
    (void)in_sizes; (void)n_in; (void)out_size; (void)ws_size;
    const float* origin = (const float*)d_in[0];
    const int*   text   = (const int*)d_in[1];
    const float* embW   = (const float*)d_in[2];
    const float* embB   = (const float*)d_in[3];
    const float* Wih0   = (const float*)d_in[4];
    const float* Whh0   = (const float*)d_in[5];
    const float* bih0   = (const float*)d_in[6];
    const float* bhh0   = (const float*)d_in[7];
    const float* Wih1   = (const float*)d_in[8];
    const float* Whh1   = (const float*)d_in[9];
    const float* bih1   = (const float*)d_in[10];
    const float* bhh1   = (const float*)d_in[11];
    const float* genW   = (const float*)d_in[12];
    const float* genB   = (const float*)d_in[13];

    char* ws = (char*)d_ws;
    unsigned short* wcat0 = (unsigned short*)(ws + OFF_WCAT0);
    unsigned short* wcat1 = (unsigned short*)(ws + OFF_WCAT1);
    unsigned short* genWb = (unsigned short*)(ws + OFF_GENW);
    unsigned short* A0    = (unsigned short*)(ws + OFF_A0);
    unsigned short* A1    = (unsigned short*)(ws + OFF_A1);
    float* h0    = (float*)(ws + OFF_H0);
    float* c0    = (float*)(ws + OFF_C0);
    float* h1    = (float*)(ws + OFF_H1);
    float* c1    = (float*)(ws + OFF_C1);
    float* gates = (float*)(ws + OFF_GATES);
    unsigned short* ctxbf = (unsigned short*)(ws + OFF_CTXBF);
    float* dgen  = (float*)(ws + OFF_DGEN);

    float* out_g       = (float*)d_out;                         // [256,32,97]
    float* out_hiddens = out_g + (size_t)ROWS * NCLS;           // [256,32,512]
    float* out_masks   = out_hiddens + (size_t)ROWS * HID;      // [256,32,8,32]

    // Zero recurrent state + A-buffers (A0|A1|h0|c0|h1|c1 are contiguous)
    hipMemsetAsync(ws + OFF_A0, 0, (size_t)(OFF_GATES - OFF_A0), stream);

    // Weight packing (every call; deterministic)
    pack_lstm_weights<<<(2048 * 1024) / 256, 256, 0, stream>>>(Wih0, Whh0, Wih1, Whh1, wcat0, wcat1);
    pack_gen_weights<<<(128 * 512) / 256, 256, 0, stream>>>(genW, genWb);

    const int ewBlocks   = (Bv * HID) / 256;                 // 512
    const int gemmBlocks = (8 * 32 * 32) / 256;              // 32  (M32=8, N64=32)

    for (int t = 0; t < NSTEP; ++t) {
        // x_emb -> A0[:, 0:512]
        emb_fill<<<ewBlocks, 256, 0, stream>>>(text, embW, embB, A0, t);
        // cell 0: gates = [x_emb | h0] @ [Wih0|Whh0]^T
        gemm_bf16_wmma<<<gemmBlocks, 256, 0, stream>>>(A0, wcat0, gates, 8, 32, KCAT, KCAT, KCAT, G4);
        // h0 bf16 -> A0[:,512:] (next-step recurrence) and A1[:,0:512] (cell1 input)
        lstm_ew<<<ewBlocks, 256, 0, stream>>>(gates, bih0, bhh0, h0, c0, A0 + HID, A1);
        // cell 1: gates = [h0 | h1] @ [Wih1|Whh1]^T
        gemm_bf16_wmma<<<gemmBlocks, 256, 0, stream>>>(A1, wcat1, gates, 8, 32, KCAT, KCAT, KCAT, G4);
        lstm_ew<<<ewBlocks, 256, 0, stream>>>(gates, bih1, bhh1, h1, c1, A1 + HID, nullptr);
        // attention + context (writes masks + output_hiddens + bf16 contexts)
        attn_ctx<<<Bv, 256, 0, stream>>>(origin, h1, out_masks, out_hiddens, ctxbf, t);
    }

    // g = contexts @ gen_W^T + gen_b   (M=8192, N=128 padded, K=512)
    gemm_bf16_wmma<<<(256 * 2 * 32) / 256, 256, 0, stream>>>(ctxbf, genWb, dgen, 256, 2, HID, HID, HID, NGEN);
    gen_epilogue<<<(ROWS * NCLS + 255) / 256, 256, 0, stream>>>(dgen, genB, out_g);
}
